// TPoseHuman_55130200212271
// MI455X (gfx1250) — matrix operand order repaired and checked by reading, not compiled, not run on hardware
//
#include <hip/hip_runtime.h>
#include <hip/hip_bf16.h>

typedef _Float16 v16h __attribute__((ext_vector_type(16)));
typedef _Float16 v8h  __attribute__((ext_vector_type(8)));
typedef float    v8f  __attribute__((ext_vector_type(8)));

#define ROWSTRIDE 160   // halves per activation row (covers padded 155 for Wv input)
#define TILE_HALVES 512 // 32 lanes * 16 halves per B tile

// ---------------------------------------------------------------------------
// Prepack a weight matrix W (rows x ld, using validCols of each row) into the
// CDNA5 WMMA f16 B-matrix fragment layout:
//   B tile (kchunk, ntile): lane l holds column N = l%16,
//   halves h=0..15 hold K = kc*32 + h + 8*((h>=8) + (l>=16)).
// Tiles stored kc-minor: tile = ntile*nK + kc, 512 halves per tile.
// ---------------------------------------------------------------------------
__global__ void prepack_kernel(const float* __restrict__ W, _Float16* __restrict__ wsH,
                               int baseTile, int rows, int validCols, int ld,
                               int nK, int nTiles) {
  int tid  = blockIdx.x * blockDim.x + threadIdx.x;
  int tile = tid >> 5;
  if (tile >= nTiles * nK) return;
  int lane  = tid & 31;
  int ntile = tile / nK;
  int kc    = tile % nK;
  int n     = ntile * 16 + (lane & 15);
  int kBase = kc * 32 + ((lane >> 4) << 3);
  _Float16* dst = wsH + (size_t)(baseTile + tile) * TILE_HALVES + lane * 16;
#pragma unroll
  for (int h = 0; h < 16; ++h) {
    int k = kBase + h + ((h >> 3) << 3);
    float v = 0.0f;
    if (n < rows && k < validCols) v = W[(size_t)n * ld + k];
    dst[h] = (_Float16)v;
  }
}

// c_latent[n] = bl[n] + sum_k Wl[n, 128+k] * nf_latent[ind][k]   (constant over points)
__global__ void latent_kernel(const float* __restrict__ Wl, const float* __restrict__ bl,
                              const float* __restrict__ nf_latent, const int* __restrict__ ind,
                              float* __restrict__ cLat) {
  int n = threadIdx.x;
  const float* lat = nf_latent + (size_t)ind[0] * 128;
  float s = bl[n];
#pragma unroll 8
  for (int k = 0; k < 128; ++k) s += Wl[(size_t)n * 256 + 128 + k] * lat[k];
  cLat[n] = s;
}

__device__ __forceinline__ void wave_wait_lds() {
  asm volatile("s_wait_dscnt 0" ::: "memory");
}

// ---------------------------------------------------------------------------
// One MLP layer for one wave's 16 points:
//   in  : LDS rows [point][feature] f16 (NK*32 features, padded with zeros)
//   out : LDS rows [point][feature] f16 (NT*16 features)
// A-fragment map (ISA 7.12.2, f16 16x32): lane holds row M = lane%16,
//   halves h: K = h + 8*((h>=8) + (lane>=16))  -> two contiguous 8-half runs.
// D map (f32 16x16): lane holds col N = lane%16, VGPR v holds row M = v + 8*(lane>=16).
// ---------------------------------------------------------------------------
template <int NK, int NT, bool RELU>
__device__ __forceinline__ void run_layer(const _Float16* __restrict__ inBuf,
                                          _Float16* __restrict__ outBuf,
                                          const _Float16* __restrict__ wsH, int baseTile,
                                          const float* __restrict__ bias, int lane) {
  const int r  = lane & 15;
  const int hh = lane >> 4;
  v16h a[NK];
#pragma unroll
  for (int kc = 0; kc < NK; ++kc) {
    const _Float16* src = inBuf + r * ROWSTRIDE + kc * 32 + hh * 8;
    v8h lo = *(const v8h*)(src);
    v8h hi = *(const v8h*)(src + 16);
    a[kc] = __builtin_shufflevector(lo, hi, 0,1,2,3,4,5,6,7,8,9,10,11,12,13,14,15);
  }
#pragma unroll
  for (int nt = 0; nt < NT; ++nt) {
    float b0 = bias[nt * 16 + r];
    v8f c;
#pragma unroll
    for (int i = 0; i < 8; ++i) c[i] = b0;
#pragma unroll
    for (int kc = 0; kc < NK; ++kc) {
      const v16h bm = *(const v16h*)(wsH + (size_t)(baseTile + nt * NK + kc) * TILE_HALVES + lane * 16);
      c = __builtin_amdgcn_wmma_f32_16x16x32_f16(false, a[kc], false, bm, (short)0, c, false, false);
    }
#pragma unroll
    for (int v = 0; v < 8; ++v) {
      float x = c[v];
      if (RELU) x = fmaxf(x, 0.0f);
      outBuf[(hh * 8 + v) * ROWSTRIDE + nt * 16 + r] = (_Float16)x;
    }
  }
}

// ---------------------------------------------------------------------------
// Main fused kernel: 4 waves/block, each wave drives 16 points through
// triplane sampling + full MLP using WMMA with LDS ping-pong.
// ---------------------------------------------------------------------------
__global__ void __launch_bounds__(128)
tpose_main(const float* __restrict__ part_coord, const float* __restrict__ viewdir,
           const float* __restrict__ part_bounds, const float* __restrict__ triplanes,
           const float* __restrict__ b1, const float* __restrict__ b2,
           const float* __restrict__ Wa, const float* __restrict__ ba,
           const float* __restrict__ bf_, const float* __restrict__ bv,
           const float* __restrict__ Wr, const float* __restrict__ br,
           const int* __restrict__ part_idx,
           const _Float16* __restrict__ wsH, const float* __restrict__ cLat,
           float* __restrict__ out, int Ntot) {
  __shared__ __align__(32) _Float16 lds[4][2][16 * ROWSTRIDE];
  const int lane = threadIdx.x & 31;
  const int wave = threadIdx.x >> 5;
  const int r    = lane & 15;
  const int hh   = lane >> 4;
  const int gp   = blockIdx.x * 64 + wave * 16 + r;   // point this (lane%16) owns

  _Float16* buf0 = lds[wave][0];
  _Float16* buf1 = lds[wave][1];

  // ---------------- Phase A: triplane bilinear sampling -> buf0[0..71] -------
  float p0 = part_coord[gp * 3 + 0];
  float p1 = part_coord[gp * 3 + 1];
  float p2 = part_coord[gp * 3 + 2];
  int pid = part_idx[gp];
  const float* bnd = part_bounds + pid * 6;   // [dim][mn,mx]
  float g[3];
#pragma unroll
  for (int d = 0; d < 3; ++d) {
    float mn = bnd[d * 2 + 0], mx = bnd[d * 2 + 1];
    float ctr = 0.5f * (mn + mx), hw = 0.5f * (mx - mn);
    float pv = (d == 0) ? p0 : ((d == 1) ? p1 : p2);
    g[d] = (pv - ctr) / hw;
  }
  const int ua[3] = {0, 0, 1};
  const int va[3] = {1, 2, 2};
#pragma unroll
  for (int k = 0; k < 3; ++k) {
    float u = g[ua[k]], v = g[va[k]];
    float px = fminf(fmaxf((u + 1.0f) * 0.5f * 127.0f, 0.0f), 127.0f);
    float py = fminf(fmaxf((v + 1.0f) * 0.5f * 127.0f, 0.0f), 127.0f);
    float x0f = floorf(px), y0f = floorf(py);
    int x0 = (int)x0f, y0 = (int)y0f;
    int x1 = min(x0 + 1, 127), y1 = min(y0 + 1, 127);
    float wx = px - x0f, wy = py - y0f;
    float w00 = (1.0f - wx) * (1.0f - wy);
    float w01 = wx * (1.0f - wy);
    float w10 = (1.0f - wx) * wy;
    float w11 = wx * wy;
    const float* PB = triplanes + (size_t)(pid * 3 + k) * 24 * 16384;
    // this lane-half covers 12 of the 24 channels of this plane
#pragma unroll 4
    for (int c = 0; c < 12; ++c) {
      int cc = hh * 12 + c;
      const float* P = PB + (size_t)cc * 16384;
      float f = P[y0 * 128 + x0] * w00 + P[y0 * 128 + x1] * w01 +
                P[y1 * 128 + x0] * w10 + P[y1 * 128 + x1] * w11;
      buf0[r * ROWSTRIDE + k * 24 + cc] = (_Float16)f;
    }
  }
  if (hh == 0) {
    // view positional encoding -> buf0 cols 128..154 (pad 155..159 with 0).
    // These columns are never touched by L2/Wl outputs (cols 0..127).
    float d0 = viewdir[gp * 3 + 0], d1 = viewdir[gp * 3 + 1], d2 = viewdir[gp * 3 + 2];
    _Float16* row = buf0 + r * ROWSTRIDE;
    row[128] = (_Float16)d0; row[129] = (_Float16)d1; row[130] = (_Float16)d2;
#pragma unroll
    for (int i = 0; i < 4; ++i) {
      float fr = (float)(1 << i);
      row[131 + 6 * i + 0] = (_Float16)__sinf(fr * d0);
      row[131 + 6 * i + 1] = (_Float16)__sinf(fr * d1);
      row[131 + 6 * i + 2] = (_Float16)__sinf(fr * d2);
      row[134 + 6 * i + 0] = (_Float16)__cosf(fr * d0);
      row[134 + 6 * i + 1] = (_Float16)__cosf(fr * d1);
      row[134 + 6 * i + 2] = (_Float16)__cosf(fr * d2);
    }
#pragma unroll
    for (int c = 155; c < 160; ++c) row[c] = (_Float16)0.0f;
  } else {
    // zero-pad feature cols 72..95 (K-pad for layer 1)
    _Float16* row = buf0 + r * ROWSTRIDE;
#pragma unroll
    for (int c = 72; c < 96; ++c) row[c] = (_Float16)0.0f;
  }
  wave_wait_lds();

  // ---------------- MLP chain (all per-wave, WMMA) ---------------------------
  run_layer<3, 8, true >(buf0, buf1, wsH,   0, b1,  lane);  wave_wait_lds(); // 72->128 relu
  run_layer<4, 8, true >(buf1, buf0, wsH,  24, b2,  lane);  wave_wait_lds(); // 128->128 relu

  // alpha head: per-point dot(net, Wa) split across lane halves
  {
    float pa = 0.0f;
#pragma unroll 8
    for (int k2 = 0; k2 < 64; ++k2) {
      int k = hh * 64 + k2;
      pa += (float)buf0[r * ROWSTRIDE + k] * Wa[k];
    }
    pa += __shfl_down(pa, 16, 32);
    if (hh == 0) out[gp] = pa + ba[0];
  }

  run_layer<4, 8, false>(buf0, buf1, wsH,  56, bf_, lane);  wave_wait_lds(); // 128->128 feature
  run_layer<4, 8, false>(buf1, buf0, wsH,  88, cLat, lane); wave_wait_lds(); // +latent const (incl bl)
  run_layer<5, 4, true >(buf0, buf1, wsH, 120, bv,  lane);  wave_wait_lds(); // 155(->160)->64 relu

  // rgb head: 3 dots of length 64, split across lane halves
#pragma unroll
  for (int ch = 0; ch < 3; ++ch) {
    float pr = 0.0f;
#pragma unroll 8
    for (int k2 = 0; k2 < 32; ++k2) {
      int k = hh * 32 + k2;
      pr += (float)buf1[r * ROWSTRIDE + k] * Wr[ch * 64 + k];
    }
    pr += __shfl_down(pr, 16, 32);
    if (hh == 0) out[Ntot + ch * Ntot + gp] = pr + br[ch];
  }
}

extern "C" void kernel_launch(void* const* d_in, const int* in_sizes, int n_in,
                              void* d_out, int out_size, void* d_ws, size_t ws_size,
                              hipStream_t stream) {
  const float* part_coord  = (const float*)d_in[0];
  const float* viewdir     = (const float*)d_in[1];
  const float* part_bounds = (const float*)d_in[2];
  const float* triplanes   = (const float*)d_in[3];
  const float* nf_latent   = (const float*)d_in[4];
  const float* W1 = (const float*)d_in[5];
  const float* b1 = (const float*)d_in[6];
  const float* W2 = (const float*)d_in[7];
  const float* b2 = (const float*)d_in[8];
  const float* Wa = (const float*)d_in[9];
  const float* ba = (const float*)d_in[10];
  const float* Wf = (const float*)d_in[11];
  const float* bf = (const float*)d_in[12];
  const float* Wl = (const float*)d_in[13];
  const float* bl = (const float*)d_in[14];
  const float* Wv = (const float*)d_in[15];
  const float* bv = (const float*)d_in[16];
  const float* Wr = (const float*)d_in[17];
  const float* br = (const float*)d_in[18];
  const int* part_idx = (const int*)d_in[19];
  const int* ind      = (const int*)d_in[20];

  _Float16* wsH = (_Float16*)d_ws;
  float* cLat = (float*)((char*)d_ws + (size_t)140 * TILE_HALVES * sizeof(_Float16));

  // Prepack weights into WMMA B-fragment layout (tile bases: 0,24,56,88,120)
  auto pack = [&](const float* W, int base, int rows, int cols, int ld, int nK, int nT) {
    int threads = nT * nK * 32;
    prepack_kernel<<<(threads + 127) / 128, 128, 0, stream>>>(W, wsH, base, rows, cols, ld, nK, nT);
  };
  pack(W1,   0, 128,  72,  72, 3, 8);   // 24 tiles
  pack(W2,  24, 128, 128, 128, 4, 8);   // 32 tiles
  pack(Wf,  56, 128, 128, 128, 4, 8);   // 32 tiles
  pack(Wl,  88, 128, 128, 256, 4, 8);   // 32 tiles (first 128 cols only)
  pack(Wv, 120,  64, 155, 155, 5, 4);   // 20 tiles

  latent_kernel<<<1, 128, 0, stream>>>(Wl, bl, nf_latent, ind, cLat);

  const int N = 262144;
  tpose_main<<<N / 64, 128, 0, stream>>>(part_coord, viewdir, part_bounds, triplanes,
                                         b1, b2, Wa, ba, bf, bv, Wr, br, part_idx,
                                         wsH, cLat, (float*)d_out, N);
}
